// MemorySelfAttention_40913858461715
// MI455X (gfx1250) — matrix operand order repaired and checked by reading, not compiled, not run on hardware
//
#include <hip/hip_runtime.h>
#include <hip/hip_bf16.h>

typedef __attribute__((ext_vector_type(16))) __bf16       v16bf;
typedef __attribute__((ext_vector_type(8)))  __bf16       v8bf;
typedef __attribute__((ext_vector_type(8)))  float        v8f;
typedef __attribute__((ext_vector_type(8)))  unsigned int v8u;

constexpr int kB = 4, kT = 1024, kE = 512, kH = 8, kDH = 64;
constexpr int kKV = 2304;                 // 256 (cmem) + 1024 (mem) + 1024 (x)
constexpr float kSCALE = 0.125f;          // 64^-0.5

// ---- flat f32 output layout (logits, new_mem, new_cmem, aux, weights) ----
constexpr size_t OUT_LOGITS  = 0;
constexpr size_t OUT_NEWMEM  = (size_t)kB * kT * kE;               // 2097152
constexpr size_t OUT_NEWCMEM = OUT_NEWMEM + (size_t)kB * kT * kE;  // 4194304
constexpr size_t OUT_AUX     = OUT_NEWCMEM + (size_t)kB * 256 * kE;// 4718592
constexpr size_t OUT_WEIGHTS = OUT_AUX + 1;

// ---- WMMA fragment helpers (contiguous k = 16*half + s order, same for A and B) ----
// A 16x32: lane 0-15 = row M; per lane 16 consecutive K values [half*16,+16)
__device__ __forceinline__ v16bf load_frag_a(const __bf16* krow, int half) {
  v8bf lo = *(const v8bf*)(krow + half * 16);
  v8bf hi = *(const v8bf*)(krow + half * 16 + 8);
  return __builtin_shufflevector(lo, hi, 0, 1, 2, 3, 4, 5, 6, 7,
                                 8, 9, 10, 11, 12, 13, 14, 15);
}
// B 32x16: lane = column N; 16 consecutive K values [half*16,+16)  -> needs B^T storage
__device__ __forceinline__ v16bf load_frag_bt(const __bf16* nrow, int half) {
  return *(const v16bf*)(nrow + half * 16);
}

// ======================= elementwise / repack kernels =======================
// f32 (Bn,L,C) -> bf16 rows placed at roff inside (Bn,dstL,C); 8 elems/thread (C%8==0)
__global__ void pack_rows_kernel(const float* __restrict__ src, __bf16* __restrict__ dst,
                                 int Bn, int L, int C, int dstL, int roff) {
  size_t idx = ((size_t)blockIdx.x * 256 + threadIdx.x) * 8;
  size_t total = (size_t)Bn * L * C;
  if (idx >= total) return;
  int c = (int)(idx % C);
  int j = (int)((idx / C) % L);
  int b = (int)(idx / ((size_t)C * L));
  const float* s = src + idx;
  __bf16* d = dst + ((size_t)b * dstL + roff + j) * C + c;
#pragma unroll
  for (int i = 0; i < 8; ++i) d[i] = (__bf16)s[i];
}

// f32 (K,N) -> bf16 (N,K)   (B^T layout for WMMA B fragments)
__global__ void pack_transpose_kernel(const float* __restrict__ src, __bf16* __restrict__ dst,
                                      int K, int N) {
  size_t idx = (size_t)blockIdx.x * 256 + threadIdx.x;
  if (idx >= (size_t)K * N) return;
  int k = (int)(idx % K);
  int n = (int)(idx / K);
  dst[idx] = (__bf16)src[(size_t)k * N + n];
}

__global__ void copy_f32_kernel(const float* __restrict__ src, float* __restrict__ dst, size_t n) {
  size_t i = ((size_t)blockIdx.x * 256 + threadIdx.x) * 4;
  if (i >= n) return;
#pragma unroll
  for (int t = 0; t < 4; ++t) dst[i + t] = src[i + t];
}

// conv_w (E,E,4)[o][e][k] -> W2^T (E, 4E): W2t[o][k*E+e]  (N=E rows, K=4E contiguous)
__global__ void repack_convw_t_kernel(const float* __restrict__ cw, __bf16* __restrict__ W2t) {
  size_t idx = (size_t)blockIdx.x * 256 + threadIdx.x;
  if (idx >= (size_t)kE * 4 * kE) return;
  int kk = (int)(idx % (4 * kE));
  int o  = (int)(idx / (4 * kE));
  int k = kk / kE, e = kk % kE;
  W2t[idx] = (__bf16)cw[((size_t)o * kE + e) * 4 + k];
}

// (B,L,H*64[+off]) -> (B,H,L,64)
__global__ void split_heads_kernel(const __bf16* __restrict__ src, __bf16* __restrict__ dst,
                                   int L, int src_ld, int col_off) {
  size_t idx = (size_t)blockIdx.x * 256 + threadIdx.x;
  size_t total = (size_t)kB * kH * L * kDH;
  if (idx >= total) return;
  int d = (int)(idx & 63);
  size_t r = idx >> 6;
  int j = (int)(r % L);
  size_t bh = r / L;
  int h = (int)(bh % kH);
  int b = (int)(bh / kH);
  dst[idx] = src[((size_t)b * L + j) * src_ld + col_off + h * kDH + d];
}

// (B,L,H*64[+off]) -> (B,H,64,L)   (transposed: B^T layout for PV GEMM / aux)
__global__ void split_heads_t_kernel(const __bf16* __restrict__ src, __bf16* __restrict__ dst,
                                     int L, int src_ld, int col_off) {
  size_t idx = (size_t)blockIdx.x * 256 + threadIdx.x;
  size_t total = (size_t)kB * kH * kDH * L;
  if (idx >= total) return;
  int j = (int)(idx % L);
  size_t t = idx / L;
  int d = (int)(t & 63);
  size_t bh = t >> 6;
  int h = (int)(bh % kH);
  int b = (int)(bh / kH);
  dst[idx] = src[((size_t)b * L + j) * src_ld + col_off + h * kDH + d];
}

__global__ void zero_kernel(float* p) { *p = 0.0f; }

// ======================= generic batched bf16 WMMA GEMM =======================
// C[z] = A[z] (MxK, row-major) * B[z] (KxN, provided TRANSPOSED as Bt (N,K)) (+bias)
// 64x64x32 tiles, 128 threads (4 waves); wave w owns rows [16w,16w+16) x 64 cols.
// Requires M%64==0, N%64==0, K%32==0 (true for every call below).
template <int A_F32, int OUT_BF16, int HAS_BIAS>
__global__ __launch_bounds__(128)
void gemm_bf16_wmma_kernel(const void* __restrict__ Ap, const __bf16* __restrict__ Btp,
                           void* __restrict__ Cp, const float* __restrict__ bias,
                           int M, int N, int K, int lda, int ldbt, int ldc,
                           long long a_bstride, long long b_bstride,
                           int c_bdiv, long long c_bstride0, long long c_bstride1) {
  __shared__ alignas(128) __bf16 As[64][32];
  __shared__ alignas(128) __bf16 Bts[64][32];
  const float*  Af = (const float*)Ap;
  const __bf16* Ab = (const __bf16*)Ap;
  int z = blockIdx.z;
  size_t aoff = (size_t)z * a_bstride;
  size_t boff = (size_t)z * b_bstride;
  size_t coff = (size_t)(z / c_bdiv) * c_bstride0 + (size_t)(z % c_bdiv) * c_bstride1;
  int row0 = blockIdx.y * 64, col0 = blockIdx.x * 64;
  int tid = threadIdx.x;
  int wave = tid >> 5, lane = tid & 31;
  int half = lane >> 4, nc = lane & 15;

  v8f acc[4] = {};

  int ar = tid >> 1, ac = (tid & 1) * 16;  // 2 thr/row, 16 contiguous elems each

  for (int k0 = 0; k0 < K; k0 += 32) {
    size_t abase = aoff + (size_t)(row0 + ar) * lda + (k0 + ac);
    size_t bbase = boff + (size_t)(col0 + ar) * ldbt + (k0 + ac);
    if (A_F32) {  // f32 attn weights -> bf16 on the fly (cvt_pk)
      const float* s = Af + abase;
#pragma unroll
      for (int i = 0; i < 16; ++i) As[ar][ac + i] = (__bf16)s[i];
    } else {      // 16B vector copies
      const v8bf* s = (const v8bf*)(Ab + abase);
      *(v8bf*)&As[ar][ac]     = s[0];
      *(v8bf*)&As[ar][ac + 8] = s[1];
    }
    {
      const v8bf* s = (const v8bf*)(Btp + bbase);
      *(v8bf*)&Bts[ar][ac]     = s[0];
      *(v8bf*)&Bts[ar][ac + 8] = s[1];
    }
    if (k0 + 32 < K) {  // CDNA5 global_prefetch_b8 of next K-tile
      __builtin_prefetch(&Btp[bbase + 32], 0, 1);
      if (A_F32) __builtin_prefetch(&Af[abase + 32], 0, 1);
      else       __builtin_prefetch(&Ab[abase + 32], 0, 1);
    }
    __syncthreads();
    v16bf a = load_frag_a(&As[wave * 16 + nc][0], half);
#pragma unroll
    for (int t = 0; t < 4; ++t) {
      v16bf bm = load_frag_bt(&Bts[t * 16 + nc][0], half);
      acc[t] = __builtin_amdgcn_wmma_f32_16x16x32_bf16(false, a, false, bm,
                                                       (short)0, acc[t], false, false);
    }
    __syncthreads();
  }
#pragma unroll
  for (int t = 0; t < 4; ++t) {
#pragma unroll
    for (int g = 0; g < 8; ++g) {
      int m = row0 + wave * 16 + g + 8 * half;  // C layout: VGPR g -> M = g / g+8
      int n = col0 + t * 16 + nc;
      float v = acc[t][g];
      if (HAS_BIAS) v += bias[n];
      size_t ci = coff + (size_t)m * ldc + n;
      if (OUT_BF16) ((__bf16*)Cp)[ci] = (__bf16)v;
      else          ((float*)Cp)[ci]  = v;
    }
  }
}

// ======================= attention scores (content + rel-pos) =======================
// Per block (128 thr = 4 waves): (b,h), 16 query rows r0.., 64 kv cols.
// wave w: one q@k^T 16x16 tile + two q@pos^T tiles covering the j' = c+1023-r band.
// k_b (j,dh) is already the B^T layout; pos fragments are 32B contiguous global loads
// masked with an integer AND (no f32 round-trip).
__global__ __launch_bounds__(128)
void attn_scores_kernel(const __bf16* __restrict__ qb,   // (B,H,T,64)
                        const __bf16* __restrict__ kb,   // (B,H,KV,64)
                        const __bf16* __restrict__ posb, // (H,KV,64)
                        float* __restrict__ wts) {       // (B,H,T,KV)
  __shared__ alignas(128) __bf16 qs[16][64];
  __shared__ alignas(128) __bf16 ks[64][64];
  __shared__ float Ps[4][16][32];
  int bh = blockIdx.z;
  int h = bh & (kH - 1);
  int r0 = blockIdx.y * 16;
  int c0base = blockIdx.x * 64;
  int tid = threadIdx.x;
  int wave = tid >> 5, lane = tid & 31;
  int half = lane >> 4, nc = lane & 15;

  const __bf16* qsrc = qb + ((size_t)bh * kT + r0) * kDH;
  for (int i = tid * 8; i < 16 * 64; i += 128 * 8)
    *(v8bf*)&qs[i >> 6][i & 63] = *(const v8bf*)(qsrc + i);
  const __bf16* ksrc = kb + ((size_t)bh * kKV + c0base) * kDH;
  for (int i = tid * 8; i < 64 * 64; i += 128 * 8)
    *(v8bf*)&ks[i >> 6][i & 63] = *(const v8bf*)(ksrc + i);
  __syncthreads();

  int c0 = c0base + wave * 16;
  int j0 = c0 - r0 + (kT - 16);            // j' band start; always >= 0
  const __bf16* ph = posb + (size_t)h * kKV * kDH;

  v8f s = {}, p0 = {}, p1 = {};
#pragma unroll
  for (int kc = 0; kc < 2; ++kc) {
    v16bf a  = load_frag_a(&qs[nc][kc * 32], half);
    v16bf bk = load_frag_bt(&ks[wave * 16 + nc][kc * 32], half);
    s = __builtin_amdgcn_wmma_f32_16x16x32_bf16(false, a, false, bk, (short)0, s, false, false);
#pragma unroll
    for (int pt = 0; pt < 2; ++pt) {
      int j = j0 + pt * 16 + nc;
      int jc = j < kKV ? j : kKV - 1;
      unsigned msk = (j < kKV) ? 0xFFFFFFFFu : 0u;  // zero-fill rows past the tensor
      const v8u* ps = (const v8u*)(const void*)(ph + (size_t)jc * kDH + kc * 32);
      v8u raw = ps[half];
      raw &= msk;
      v16bf bp = __builtin_bit_cast(v16bf, raw);
      if (pt == 0)
        p0 = __builtin_amdgcn_wmma_f32_16x16x32_bf16(false, a, false, bp, (short)0, p0, false, false);
      else
        p1 = __builtin_amdgcn_wmma_f32_16x16x32_bf16(false, a, false, bp, (short)0, p1, false, false);
    }
  }
  // spill P (16x32) through LDS for the cross-lane diagonal gather
#pragma unroll
  for (int g = 0; g < 8; ++g) {
    int m = g + 8 * half;
    Ps[wave][m][nc]      = p0[g];
    Ps[wave][m][16 + nc] = p1[g];
  }
  __syncthreads();
#pragma unroll
  for (int g = 0; g < 8; ++g) {
    int rr = g + 8 * half;
    int r = r0 + rr;
    int c = c0 + nc;
    float v = s[g];
    if (c - r <= (kKV - kT)) v += Ps[wave][rr][nc + 15 - rr];  // shift(): j' = c+1023-r
    wts[((size_t)bh * kT + r) * kKV + c] = v * kSCALE;
  }
}

// ======================= row softmax (in place, KV=2304) =======================
__global__ __launch_bounds__(256)
void softmax_kernel(float* __restrict__ wts) {
  __shared__ float sh[kKV];
  __shared__ float red[256];
  float* p = wts + (size_t)blockIdx.x * kKV;
  int tid = threadIdx.x;
  float m = -3.0e38f;
  for (int j = tid; j < kKV; j += 256) { float v = p[j]; sh[j] = v; m = fmaxf(m, v); }
  red[tid] = m; __syncthreads();
  for (int s = 128; s > 0; s >>= 1) { if (tid < s) red[tid] = fmaxf(red[tid], red[tid + s]); __syncthreads(); }
  m = red[0]; __syncthreads();
  float sum = 0.0f;
  for (int j = tid; j < kKV; j += 256) { float e = __expf(sh[j] - m); sh[j] = e; sum += e; }
  red[tid] = sum; __syncthreads();
  for (int s = 128; s > 0; s >>= 1) { if (tid < s) red[tid] += red[tid + s]; __syncthreads(); }
  float inv = 1.0f / red[0];
  for (int j = tid; j < kKV; j += 256) p[j] = sh[j] * inv;
}

// ======================= aux loss: MSE(full_attn(q,om), full_attn(q,c)) =======================
// one block per (b,h,r); f32 VALU path (~10 GFLOP of the op). v is transposed (B,H,64,KV).
__global__ __launch_bounds__(256)
void aux_loss_kernel(const __bf16* __restrict__ qb, const __bf16* __restrict__ kb,
                     const __bf16* __restrict__ vtb, const __bf16* __restrict__ ckv,
                     float* __restrict__ aux) {
  __shared__ float s1[1024];
  __shared__ float qsh[64];
  __shared__ float red[256];
  __shared__ float pp[4][64];
  __shared__ float o1[64];
  __shared__ float o2[64];
  int tid = threadIdx.x;
  size_t bhr = blockIdx.x;
  int r = (int)(bhr % kT);
  size_t bh = bhr / kT;
  int h = (int)(bh % kH);
  int b = (int)(bh / kH);
  const __bf16* q = qb + (bh * kT + r) * kDH;
  if (tid < 64) qsh[tid] = (float)q[tid];
  __syncthreads();
  int d = tid & 63, part = tid >> 6;

  // --- attn over old-mem K/V: kv rows 256..1279 ---
  const __bf16* kk = kb + (bh * kKV + 256) * kDH;
  const __bf16* vt = vtb + (bh * kDH) * kKV;  // vt[d][j] at d*KV + j
  for (int j = tid; j < 1024; j += 256) {
    const __bf16* krow = kk + (size_t)j * kDH;
    float s = 0.0f;
    for (int dd = 0; dd < kDH; ++dd) s += qsh[dd] * (float)krow[dd];
    s1[j] = s * kSCALE;
  }
  __syncthreads();
  float m = -3.0e38f;
  for (int j = tid; j < 1024; j += 256) m = fmaxf(m, s1[j]);
  red[tid] = m; __syncthreads();
  for (int s = 128; s > 0; s >>= 1) { if (tid < s) red[tid] = fmaxf(red[tid], red[tid + s]); __syncthreads(); }
  m = red[0]; __syncthreads();
  float sum = 0.0f;
  for (int j = tid; j < 1024; j += 256) { float e = __expf(s1[j] - m); s1[j] = e; sum += e; }
  red[tid] = sum; __syncthreads();
  for (int s = 128; s > 0; s >>= 1) { if (tid < s) red[tid] += red[tid + s]; __syncthreads(); }
  float inv1 = 1.0f / red[0];
  __syncthreads();
  float acc = 0.0f;
  for (int j = part * 256; j < part * 256 + 256; ++j)
    acc += s1[j] * (float)vt[(size_t)d * kKV + 256 + j];   // contiguous in j
  pp[part][d] = acc; __syncthreads();
  if (tid < 64) o1[tid] = (pp[0][tid] + pp[1][tid] + pp[2][tid] + pp[3][tid]) * inv1;
  __syncthreads();

  // --- attn over compressed K/V (256 keys); ckv row = (b*256+j, 2E), k at 0, v at E ---
  {
    const __bf16* crow = ckv + ((size_t)(b * 256 + tid)) * (2 * kE) + h * kDH;
    float s = 0.0f;
    for (int dd = 0; dd < kDH; ++dd) s += qsh[dd] * (float)crow[dd];
    s1[tid] = s * kSCALE;
  }
  __syncthreads();
  red[tid] = s1[tid]; __syncthreads();
  for (int s = 128; s > 0; s >>= 1) { if (tid < s) red[tid] = fmaxf(red[tid], red[tid + s]); __syncthreads(); }
  m = red[0]; __syncthreads();
  float e2 = __expf(s1[tid] - m);
  s1[tid] = e2;
  red[tid] = e2; __syncthreads();
  for (int s = 128; s > 0; s >>= 1) { if (tid < s) red[tid] += red[tid + s]; __syncthreads(); }
  float inv2 = 1.0f / red[0];
  __syncthreads();
  float acc2 = 0.0f;
  for (int j = part * 64; j < part * 64 + 64; ++j)
    acc2 += s1[j] * (float)ckv[((size_t)(b * 256 + j)) * (2 * kE) + kE + h * kDH + d];
  pp[part][d] = acc2; __syncthreads();
  if (tid < 64) o2[tid] = (pp[0][tid] + pp[1][tid] + pp[2][tid] + pp[3][tid]) * inv2;
  __syncthreads();

  float dsq = 0.0f;
  if (tid < 64) { float df = o1[tid] - o2[tid]; dsq = df * df; }
  red[tid] = dsq; __syncthreads();
  for (int s = 128; s > 0; s >>= 1) { if (tid < s) red[tid] += red[tid + s]; __syncthreads(); }
  if (tid == 0) atomicAdd(aux, red[0] * (1.0f / 2097152.0f));  // mean over B*H*T*DH
}

// ======================= host launcher =======================
extern "C" void kernel_launch(void* const* d_in, const int* in_sizes, int n_in,
                              void* d_out, int out_size, void* d_ws, size_t ws_size,
                              hipStream_t stream) {
  const float* x     = (const float*)d_in[0];
  const float* mem   = (const float*)d_in[1];
  const float* cmem  = (const float*)d_in[2];
  const float* pos   = (const float*)d_in[3];
  const float* Wq    = (const float*)d_in[4];
  const float* Wkv   = (const float*)d_in[5];
  const float* Wout  = (const float*)d_in[6];
  const float* bout  = (const float*)d_in[7];
  const float* convw = (const float*)d_in[8];
  const float* convb = (const float*)d_in[9];
  float* out = (float*)d_out;
  (void)in_sizes; (void)n_in; (void)out_size; (void)ws_size;

  // workspace carve-out (~78 MB of bf16 staging)
  char* ws = (char*)d_ws;
  size_t off = 0;
  auto take = [&](size_t elems) {
    __bf16* p = (__bf16*)(ws + off);
    off += ((elems * sizeof(__bf16) + 255) & ~(size_t)255);
    return p;
  };
  __bf16* xb     = take((size_t)kB * kT * kE);
  __bf16* kvinb  = take((size_t)kB * kKV * kE);
  __bf16* memb   = take((size_t)kB * kT * kE);
  __bf16* posb   = take((size_t)kH * kKV * kDH);
  __bf16* Wqt    = take((size_t)kE * kE);          // (N=E, K=E)
  __bf16* Wkvt   = take((size_t)2 * kE * kE);      // (N=2E, K=E)
  __bf16* Woutt  = take((size_t)kE * kE);          // (N=E, K=E)
  __bf16* W2t    = take((size_t)kE * 4 * kE);      // (N=E, K=4E)
  __bf16* qtmpb  = take((size_t)kB * kT * kE);
  __bf16* q_b    = take((size_t)kB * kH * kT * kDH);
  __bf16* kvtmpb = take((size_t)kB * kKV * 2 * kE);
  __bf16* k_b    = take((size_t)kB * kH * kKV * kDH);   // (B,H,KV,64)
  __bf16* v_t    = take((size_t)kB * kH * kDH * kKV);   // (B,H,64,KV) transposed
  __bf16* outb   = take((size_t)kB * kT * kE);
  __bf16* compb  = take((size_t)kB * 256 * kE);
  __bf16* ckvb   = take((size_t)kB * 256 * 2 * kE);

  auto blocks  = [](size_t n) { return (unsigned)((n + 255) / 256); };
  auto blocks8 = [](size_t n) { return (unsigned)((n / 8 + 255) / 256); };

  // 1) bf16 packs / concat / transposes
  pack_rows_kernel<<<blocks8((size_t)kB*kT*kE), 256, 0, stream>>>(x, xb, kB, kT, kE, kT, 0);
  pack_rows_kernel<<<blocks8((size_t)kB*256*kE), 256, 0, stream>>>(cmem, kvinb, kB, 256, kE, kKV, 0);
  pack_rows_kernel<<<blocks8((size_t)kB*kT*kE), 256, 0, stream>>>(mem, kvinb, kB, kT, kE, kKV, 256);
  pack_rows_kernel<<<blocks8((size_t)kB*kT*kE), 256, 0, stream>>>(x, kvinb, kB, kT, kE, kKV, 1280);
  pack_rows_kernel<<<blocks8((size_t)kB*kT*kE), 256, 0, stream>>>(mem, memb, kB, kT, kE, kT, 0);
  pack_rows_kernel<<<blocks8((size_t)kH*kKV*kDH), 256, 0, stream>>>(pos, posb, 1, kH*kKV, kDH, kH*kKV, 0);
  pack_transpose_kernel<<<blocks((size_t)kE*kE), 256, 0, stream>>>(Wq, Wqt, kE, kE);
  pack_transpose_kernel<<<blocks((size_t)kE*2*kE), 256, 0, stream>>>(Wkv, Wkvt, kE, 2*kE);
  pack_transpose_kernel<<<blocks((size_t)kE*kE), 256, 0, stream>>>(Wout, Woutt, kE, kE);
  repack_convw_t_kernel<<<blocks((size_t)kE*4*kE), 256, 0, stream>>>(convw, W2t);

  // new_mem = x (queue[:, -MEM_LEN:] with MEM_LEN == T)
  copy_f32_kernel<<<blocks((size_t)kB*kT*kE/4), 256, 0, stream>>>(x, out + OUT_NEWMEM, (size_t)kB*kT*kE);

  // 2) q = x @ Wq -> head split
  gemm_bf16_wmma_kernel<0,1,0><<<dim3(kE/64, (kB*kT)/64, 1), 128, 0, stream>>>(
      xb, Wqt, qtmpb, nullptr, kB*kT, kE, kE, kE, kE, kE, 0, 0, 1, 0, 0);
  split_heads_kernel<<<blocks((size_t)kB*kH*kT*kDH), 256, 0, stream>>>(qtmpb, q_b, kT, kE, 0);

  // 3) k,v = [cmem;mem;x] @ Wkv -> k head-split, v head-split transposed
  gemm_bf16_wmma_kernel<0,1,0><<<dim3((2*kE)/64, (kB*kKV)/64, 1), 128, 0, stream>>>(
      kvinb, Wkvt, kvtmpb, nullptr, kB*kKV, 2*kE, kE, kE, kE, 2*kE, 0, 0, 1, 0, 0);
  split_heads_kernel<<<blocks((size_t)kB*kH*kKV*kDH), 256, 0, stream>>>(kvtmpb, k_b, kKV, 2*kE, 0);
  split_heads_t_kernel<<<blocks((size_t)kB*kH*kDH*kKV), 256, 0, stream>>>(kvtmpb, v_t, kKV, 2*kE, kE);

  // 4) scores (content + shifted rel-pos) -> weights region; 5) softmax in place
  attn_scores_kernel<<<dim3(kKV/64, kT/16, kB*kH), 128, 0, stream>>>(q_b, k_b, posb, out + OUT_WEIGHTS);
  softmax_kernel<<<kB*kH*kT, 256, 0, stream>>>(out + OUT_WEIGHTS);

  // 6) out = attn @ v (f32 A converted on the fly; B = v^T; head-scattered bf16 output)
  gemm_bf16_wmma_kernel<1,1,0><<<dim3(1, kT/64, kB*kH), 128, 0, stream>>>(
      out + OUT_WEIGHTS, v_t, outb, nullptr, kT, kDH, kKV, kKV, kKV, kE,
      (long long)kT * kKV, (long long)kDH * kKV, kH, (long long)kT * kE, (long long)kDH);

  // 7) logits = out @ Wout + bout -> d_out
  gemm_bf16_wmma_kernel<0,0,1><<<dim3(kE/64, (kB*kT)/64, 1), 128, 0, stream>>>(
      outb, Woutt, out + OUT_LOGITS, bout, kB*kT, kE, kE, kE, kE, kE, 0, 0, 1, 0, 0);

  // 8) new_cmem = conv(mem): (B*256, 2048) @ (2048, 512) + conv_b
  gemm_bf16_wmma_kernel<0,0,1><<<dim3(kE/64, (kB*256)/64, 1), 128, 0, stream>>>(
      memb, W2t, out + OUT_NEWCMEM, convb, kB*256, kE, 4*kE, 4*kE, 4*kE, kE, 0, 0, 1, 0, 0);
  pack_rows_kernel<<<blocks8((size_t)kB*256*kE), 256, 0, stream>>>(out + OUT_NEWCMEM, compb,
                                                                   1, kB*256, kE, kB*256, 0);

  // 9) ck,cv = compressed @ Wkv
  gemm_bf16_wmma_kernel<0,1,0><<<dim3((2*kE)/64, (kB*256)/64, 1), 128, 0, stream>>>(
      compb, Wkvt, ckvb, nullptr, kB*256, 2*kE, kE, kE, kE, 2*kE, 0, 0, 1, 0, 0);

  // 10) aux loss
  zero_kernel<<<1, 1, 0, stream>>>(out + OUT_AUX);
  aux_loss_kernel<<<kB*kH*kT, 256, 0, stream>>>(q_b, k_b, v_t, ckvb, out + OUT_AUX);
}
